// RankMessagePassingLayer_30855045055019
// MI455X (gfx1250) — matrix operand order repaired and checked by reading, not compiled, not run on hardware
//
#include <hip/hip_runtime.h>
#include <stdint.h>

// Problem constants (reference: B=64, C=729, S=243, D=512)
#define BB 64
#define CC 729
#define SS 243
#define DD 512

typedef __attribute__((ext_vector_type(16))) __bf16 v16bf;
typedef __attribute__((ext_vector_type(8)))  __bf16 v8bf;
typedef __attribute__((ext_vector_type(8)))  float  v8f;
typedef __attribute__((ext_vector_type(4)))  int    v4i;

// Use CDNA5 async global->LDS DMA when the toolchain exposes it.
#if defined(__has_builtin)
#if __has_builtin(__builtin_amdgcn_global_load_async_to_lds_b128) && \
    __has_builtin(__builtin_amdgcn_s_wait_asynccnt)
#define USE_ASYNC_LDS 1
#endif
#endif
#ifndef USE_ASYNC_LDS
#define USE_ASYNC_LDS 0
#endif

__device__ __forceinline__ unsigned short f2bf(float f) {
  unsigned int u = __float_as_uint(f);
  unsigned int r = u + 0x7FFFu + ((u >> 16) & 1u);   // round-to-nearest-even
  return (unsigned short)(r >> 16);
}

#if USE_ASYNC_LDS
__device__ __forceinline__ void cp_async_b128(const void* g, void* l) {
  // Builtin params are int4 pointers in AS1 (global) and AS3 (LDS).
  // generic->AS1 is value-preserving; a generic LDS pointer's low 32 bits
  // are the LDS offset, so truncation implements the addrspacecast.
  __builtin_amdgcn_global_load_async_to_lds_b128(
      (__attribute__((address_space(1))) v4i*)(unsigned long long)(uintptr_t)g,
      (__attribute__((address_space(3))) v4i*)(unsigned int)(uintptr_t)l,
      0, 0);
}
#endif

union FragU { v16bf v; v8bf h[2]; };

// ---------------------------------------------------------------------------
// out[M,N] = A[M,K](bf16) @ W[N,K](bf16)^T (+bias) (optional relu),
// stored as f32 (OUTB=false) or bf16 (OUTB=true).
// Block tile 128x128, BK=32, 256 threads = 8 waves (2 M-strips x 4 N-strips);
// each wave computes 64x32 via 4x2 WMMA accumulators. Double-buffered LDS,
// filled with async global->LDS DMA when available.
// ---------------------------------------------------------------------------
template<bool BIAS, bool RELU, bool OUTB>
__global__ __launch_bounds__(256)
void gemm_bf16_wmma(const unsigned short* __restrict__ Ag,
                    const unsigned short* __restrict__ Wg,
                    const float* __restrict__ bias,
                    float* __restrict__ outF,
                    unsigned short* __restrict__ outB,
                    int M, int N, int K) {
  constexpr int BM = 128, BN = 128, BK = 32;
  constexpr int LDT = BK + 8;   // padded row stride (keeps 16B alignment)
  __shared__ __align__(16) __bf16 sA[2][BM * LDT];
  __shared__ __align__(16) __bf16 sB[2][BN * LDT];

  const int t    = threadIdx.x;
  const int lane = t & 31;
  const int wave = t >> 5;
  const int r    = lane & 15;   // row/col within 16x16 tile
  const int h    = lane >> 4;   // half-wave selector
  const int wy   = wave >> 2;   // 0..1 : 64-row strip in M
  const int wx   = wave & 3;    // 0..3 : 32-col strip in N

  const int nBase = blockIdx.x * BN;
  const int mBase = blockIdx.y * BM;

  v8f acc[4][2];
#pragma unroll
  for (int i = 0; i < 4; ++i)
#pragma unroll
    for (int j = 0; j < 2; ++j)
#pragma unroll
      for (int e = 0; e < 8; ++e) acc[i][j][e] = 0.0f;

  // Global->LDS staging: each of 256 threads moves 32B of A and 32B of B.
  const int srow = t >> 1;          // 0..127
  const int skh  = (t & 1) * 16;    // 0 / 16 bf16 elements
  int garow = mBase + srow;
  if (garow >= M) garow = M - 1;    // clamp (stores masked later)
  const unsigned short* aSrc = Ag + (size_t)garow * K + skh;
  const unsigned short* bSrc = Wg + (size_t)(nBase + srow) * K + skh;
  __bf16* aStage[2] = { &sA[0][srow * LDT + skh], &sA[1][srow * LDT + skh] };
  __bf16* bStage[2] = { &sB[0][srow * LDT + skh], &sB[1][srow * LDT + skh] };

  auto stage = [&](int buf, int k0) {
    const unsigned short* a = aSrc + k0;
    const unsigned short* b = bSrc + k0;
#if USE_ASYNC_LDS
    cp_async_b128(a,     aStage[buf]);
    cp_async_b128(a + 8, aStage[buf] + 8);
    cp_async_b128(b,     bStage[buf]);
    cp_async_b128(b + 8, bStage[buf] + 8);
#else
    const uint4* as = (const uint4*)a;
    const uint4* bs = (const uint4*)b;
    uint4* ad = (uint4*)aStage[buf];
    uint4* bd = (uint4*)bStage[buf];
    ad[0] = as[0]; ad[1] = as[1];
    bd[0] = bs[0]; bd[1] = bs[1];
#endif
  };

  const int KT = K / BK;
  stage(0, 0);
  for (int kb = 0; kb < KT; ++kb) {
    const int cur = kb & 1;
#if USE_ASYNC_LDS
    __builtin_amdgcn_s_wait_asynccnt(0);   // my DMA into buf[cur] landed
#endif
    __syncthreads();                        // everyone's tile ready; prior reads done
    if (kb + 1 < KT) stage(cur ^ 1, (kb + 1) * BK);

    // A fragments (ISA 7.12.2: lane holds row M=r; K = 8h..8h+7 then 16+8h..)
    FragU af[4], bfr[2];
    const __bf16* cA = sA[cur];
    const __bf16* cB = sB[cur];
#pragma unroll
    for (int mt = 0; mt < 4; ++mt) {
      const __bf16* ap = cA + (wy * 64 + mt * 16 + r) * LDT;
      af[mt].h[0] = *(const v8bf*)(ap + 8 * h);
      af[mt].h[1] = *(const v8bf*)(ap + 16 + 8 * h);
    }
    // B fragments: lane holds column N=r; K = 8h..8h+7 then 16+8h..
#pragma unroll
    for (int nt = 0; nt < 2; ++nt) {
      const __bf16* bp = cB + (wx * 32 + nt * 16 + r) * LDT;
      bfr[nt].h[0] = *(const v8bf*)(bp + 8 * h);
      bfr[nt].h[1] = *(const v8bf*)(bp + 16 + 8 * h);
    }
#pragma unroll
    for (int mt = 0; mt < 4; ++mt)
#pragma unroll
      for (int nt = 0; nt < 2; ++nt)
        acc[mt][nt] = __builtin_amdgcn_wmma_f32_16x16x32_bf16(
            false, af[mt].v, false, bfr[nt].v, (short)0, acc[mt][nt], false, false);
  }

  // Epilogue. D layout: VGPR j -> row m0 + j + 8h, col n0 + r.
#pragma unroll
  for (int mt = 0; mt < 4; ++mt) {
#pragma unroll
    for (int nt = 0; nt < 2; ++nt) {
      const int n = nBase + wx * 32 + nt * 16 + r;
      const float bv = BIAS ? bias[n] : 0.0f;
      const int m0 = mBase + wy * 64 + mt * 16 + 8 * h;
#pragma unroll
      for (int j = 0; j < 8; ++j) {
        const int m = m0 + j;
        if (m < M) {
          float v = acc[mt][nt][j] + bv;
          if (RELU) v = v > 0.0f ? v : 0.0f;
          if (OUTB) outB[(size_t)m * N + n] = f2bf(v);
          else      outF[(size_t)m * N + n] = v;
        }
      }
    }
  }
}

// ---------------------------------------------------------------------------
__global__ void cvt_f32_bf16(const float* __restrict__ in,
                             unsigned short* __restrict__ out, long long n) {
  long long i = (long long)blockIdx.x * blockDim.x + threadIdx.x;
  const long long stride = (long long)gridDim.x * blockDim.x;
  for (; i < n; i += stride) out[i] = f2bf(in[i]);
}

// Scatter-add candidate messages into per-set accumulators (3 targets each).
__global__ __launch_bounds__(128)
void scatter_kernel(const float* __restrict__ msgs, const int* __restrict__ idx,
                    float* __restrict__ acc) {
  const int bc = blockIdx.x;
  const int b = bc / CC;
  const int c = bc - b * CC;
  const int s0 = idx[3 * c + 0];
  const int s1 = idx[3 * c + 1];
  const int s2 = idx[3 * c + 2];
  const float* m = msgs + (size_t)bc * DD;
  float* a0 = acc + ((size_t)b * SS + s0) * DD;
  float* a1 = acc + ((size_t)b * SS + s1) * DD;
  float* a2 = acc + ((size_t)b * SS + s2) * DD;
  for (int d = threadIdx.x; d < DD; d += 128) {
    const float v = m[d];
    unsafeAtomicAdd(&a0[d], v);   // global_atomic_add_f32
    unsafeAtomicAdd(&a1[d], v);
    unsafeAtomicAdd(&a2[d], v);
  }
}

// Gather 3 set rows per candidate, mean, convert to bf16 for the next GEMM.
__global__ __launch_bounds__(128)
void gather_kernel(const float* __restrict__ setOut, const int* __restrict__ idx,
                   unsigned short* __restrict__ gm) {
  const int bc = blockIdx.x;
  const int b = bc / CC;
  const int c = bc - b * CC;
  const float* p0 = setOut + ((size_t)b * SS + idx[3 * c + 0]) * DD;
  const float* p1 = setOut + ((size_t)b * SS + idx[3 * c + 1]) * DD;
  const float* p2 = setOut + ((size_t)b * SS + idx[3 * c + 2]) * DD;
  unsigned short* g = gm + (size_t)bc * DD;
  for (int d = threadIdx.x; d < DD; d += 128)
    g[d] = f2bf((p0[d] + p1[d] + p2[d]) * (1.0f / 3.0f));
}

// y = LN(x1 + scale2*x2) * gamma + beta, one block per row of D=512.
template<bool OUTB>
__global__ __launch_bounds__(256)
void ln_kernel(const float* __restrict__ x1, const float* __restrict__ x2,
               float scale2, const float* __restrict__ gamma,
               const float* __restrict__ beta, float* __restrict__ outF,
               unsigned short* __restrict__ outB) {
  __shared__ float red[256];
  __shared__ float stat[2];
  const int t = threadIdx.x;
  const size_t base = (size_t)blockIdx.x * DD;

  const float a0 = x1[base + t]       + scale2 * x2[base + t];
  const float a1 = x1[base + t + 256] + scale2 * x2[base + t + 256];

  red[t] = a0 + a1;
  __syncthreads();
  for (int off = 128; off > 0; off >>= 1) {
    if (t < off) red[t] += red[t + off];
    __syncthreads();
  }
  if (t == 0) stat[0] = red[0] * (1.0f / DD);
  __syncthreads();
  const float mean = stat[0];
  const float d0 = a0 - mean, d1 = a1 - mean;

  red[t] = d0 * d0 + d1 * d1;
  __syncthreads();
  for (int off = 128; off > 0; off >>= 1) {
    if (t < off) red[t] += red[t + off];
    __syncthreads();
  }
  if (t == 0) stat[1] = rsqrtf(red[0] * (1.0f / DD) + 1e-5f);
  __syncthreads();
  const float rstd = stat[1];

  const float y0 = d0 * rstd * gamma[t]       + beta[t];
  const float y1 = d1 * rstd * gamma[t + 256] + beta[t + 256];
  outF[base + t] = y0;
  outF[base + t + 256] = y1;
  if (OUTB) {
    outB[base + t] = f2bf(y0);
    outB[base + t + 256] = f2bf(y1);
  }
}

// ---------------------------------------------------------------------------
extern "C" void kernel_launch(void* const* d_in, const int* in_sizes, int n_in,
                              void* d_out, int out_size, void* d_ws, size_t ws_size,
                              hipStream_t stream) {
  const float* cand_feat = (const float*)d_in[0];
  const float* set_feat  = (const float*)d_in[1];
  const float* W_vc  = (const float*)d_in[2];
  const float* b_vc  = (const float*)d_in[3];
  const float* W_vs  = (const float*)d_in[4];
  const float* b_vs  = (const float*)d_in[5];
  const float* ff_W1 = (const float*)d_in[6];
  const float* ff_b1 = (const float*)d_in[7];
  const float* ff_W2 = (const float*)d_in[8];
  const float* ff_b2 = (const float*)d_in[9];
  const float* g_set   = (const float*)d_in[10];
  const float* be_set  = (const float*)d_in[11];
  const float* g_cand  = (const float*)d_in[12];
  const float* be_cand = (const float*)d_in[13];
  const float* g_ff    = (const float*)d_in[14];
  const float* be_ff   = (const float*)d_in[15];
  const int*   idx     = (const int*)d_in[16];

  const size_t M = (size_t)BB * CC;        // 46656 candidate rows
  const size_t RS = (size_t)BB * SS;       // 15552 set rows

  char* ws = (char*)d_ws;
  size_t off = 0;
  auto alloc = [&](size_t bytes) {
    size_t o = off;
    off += (bytes + 255) & ~(size_t)255;
    return o;
  };
  unsigned short* candB  = (unsigned short*)(ws + alloc(M * DD * 2));
  unsigned short* wvcB   = (unsigned short*)(ws + alloc((size_t)DD * DD * 2));
  unsigned short* wvsB   = (unsigned short*)(ws + alloc((size_t)DD * DD * 2));
  unsigned short* w1B    = (unsigned short*)(ws + alloc((size_t)2 * DD * DD * 2));
  unsigned short* w2B    = (unsigned short*)(ws + alloc((size_t)2 * DD * DD * 2));
  float*          bigF   = (float*)(ws + alloc(M * DD * 4));   // msgs -> upd -> ff2
  float*          setAcc = (float*)(ws + alloc(RS * DD * 4));
  unsigned short* gmB    = (unsigned short*)(ws + alloc(M * DD * 2));
  float*          candHF = (float*)(ws + alloc(M * DD * 4));
  unsigned short* candHB = (unsigned short*)(ws + alloc(M * DD * 2));
  unsigned short* h1B    = (unsigned short*)(ws + alloc(M * 2 * DD * 2));

  float* outCand = (float*)d_out;                 // (B,C,D) f32
  float* outSet  = (float*)d_out + M * DD;        // (B,S,D) f32

  const dim3 blk(256);
  const dim3 gN512(DD / 128, (unsigned)((M + 127) / 128));
  const dim3 gN1024(2 * DD / 128, (unsigned)((M + 127) / 128));

  // bf16 conversions
  cvt_f32_bf16<<<2048, 256, 0, stream>>>(cand_feat, candB, (long long)(M * DD));
  cvt_f32_bf16<<<256, 256, 0, stream>>>(W_vc,  wvcB, (long long)DD * DD);
  cvt_f32_bf16<<<256, 256, 0, stream>>>(W_vs,  wvsB, (long long)DD * DD);
  cvt_f32_bf16<<<512, 256, 0, stream>>>(ff_W1, w1B,  (long long)2 * DD * DD);
  cvt_f32_bf16<<<512, 256, 0, stream>>>(ff_W2, w2B,  (long long)2 * DD * DD);

  // 1) cand_msgs = cand @ W_vc^T + b_vc
  gemm_bf16_wmma<true, false, false><<<gN512, blk, 0, stream>>>(
      candB, wvcB, b_vc, bigF, nullptr, (int)M, DD, DD);

  // 2) scatter-add into set accumulators, then set LN
  (void)hipMemsetAsync(setAcc, 0, RS * DD * 4, stream);
  scatter_kernel<<<(unsigned)M, 128, 0, stream>>>(bigF, idx, setAcc);
  ln_kernel<false><<<(unsigned)RS, 256, 0, stream>>>(
      set_feat, setAcc, 1.0f / 9.0f, g_set, be_set, outSet, nullptr);

  // 3) gather mean -> GEMM with W_vs -> cand LN
  gather_kernel<<<(unsigned)M, 128, 0, stream>>>(outSet, idx, gmB);
  gemm_bf16_wmma<true, false, false><<<gN512, blk, 0, stream>>>(
      gmB, wvsB, b_vs, bigF, nullptr, (int)M, DD, DD);
  ln_kernel<true><<<(unsigned)M, 256, 0, stream>>>(
      cand_feat, bigF, 1.0f, g_cand, be_cand, candHF, candHB);

  // 4) FF block: relu(cand_h @ W1^T + b1) @ W2^T + b2, then final LN
  gemm_bf16_wmma<true, true, true><<<gN1024, blk, 0, stream>>>(
      candHB, w1B, ff_b1, nullptr, h1B, (int)M, 2 * DD, DD);
  gemm_bf16_wmma<true, false, false><<<gN512, blk, 0, stream>>>(
      h1B, w2B, ff_b2, bigF, nullptr, (int)M, DD, 2 * DD);
  ln_kernel<false><<<(unsigned)M, 256, 0, stream>>>(
      candHF, bigF, 1.0f, g_ff, be_ff, outCand, nullptr);
}
